// PointNetSetAbstraction_63462436766094
// MI455X (gfx1250) — matrix operand order repaired and checked by reading, not compiled, not run on hardware
//
#include <hip/hip_runtime.h>
#include <hip/hip_fp16.h>

typedef __attribute__((ext_vector_type(16))) _Float16 v16h;
typedef __attribute__((ext_vector_type(8)))  _Float16 v8h;
typedef __attribute__((ext_vector_type(8)))  float    v8f;

#define BB    8
#define NN    4160
#define DD    64
#define SS    1024
#define NSAMP 32
#define PP    (BB*SS*NSAMP)   /* 262144 rows */
#define K0PAD 96              /* 67 -> 96 (3 k-tiles of 32) */
#define RAD2  0.25f           /* 0.5^2 */

// ---------------------------------------------------------------------------
// 1) Farthest point sampling: one workgroup per batch, distance array in LDS,
//    block-wide argmax each of the 1024 sequential iterations.
// ---------------------------------------------------------------------------
__global__ __launch_bounds__(256) void fps_kernel(const float* __restrict__ xyz,
                                                  float* __restrict__ newxyz) {
    __shared__ float dist[NN];
    __shared__ float rv[256];
    __shared__ int   ri[256];
    const int b   = blockIdx.x;
    const int tid = threadIdx.x;
    const float* X = xyz + (size_t)b * NN * 3;

    for (int i = tid; i < NN; i += 256) dist[i] = 1.0e10f;
    __syncthreads();

    int far = 0;
    for (int it = 0; it < SS; ++it) {
        // record centroid BEFORE the update (matches the scan's carry output)
        float cx = X[far*3+0], cy = X[far*3+1], cz = X[far*3+2];
        if (tid == 0) {
            int o = (b*SS + it) * 3;
            newxyz[o+0] = cx; newxyz[o+1] = cy; newxyz[o+2] = cz;
        }
        float bv = -1.0f; int bi2 = 0;
        for (int i = tid; i < NN; i += 256) {
            float dx = X[i*3+0]-cx, dy = X[i*3+1]-cy, dz = X[i*3+2]-cz;
            float d  = dx*dx + dy*dy + dz*dz;
            float nd = fminf(dist[i], d);
            dist[i] = nd;
            if (nd > bv) { bv = nd; bi2 = i; }
        }
        rv[tid] = bv; ri[tid] = bi2;
        __syncthreads();
        for (int s2 = 128; s2 > 0; s2 >>= 1) {
            if (tid < s2) {
                float ov = rv[tid+s2]; int oi = ri[tid+s2];
                if (ov > rv[tid] || (ov == rv[tid] && oi < ri[tid])) { rv[tid] = ov; ri[tid] = oi; }
            }
            __syncthreads();
        }
        far = ri[0];
        __syncthreads();
    }
}

// ---------------------------------------------------------------------------
// 2) Ball query: one thread per centroid, nearest-32 insertion list +
//    radius fallback to the closest point.
// ---------------------------------------------------------------------------
__global__ __launch_bounds__(128) void ball_kernel(const float* __restrict__ xyz,
                                                   const float* __restrict__ newxyz,
                                                   int* __restrict__ gidx) {
    int s = blockIdx.x * 128 + threadIdx.x;   // 0 .. BB*SS-1
    if (s >= BB*SS) return;
    int b = s / SS;
    const float* q = newxyz + (size_t)s * 3;
    float qx = q[0], qy = q[1], qz = q[2];
    const float* X = xyz + (size_t)b * NN * 3;

    float bd[NSAMP]; int bi[NSAMP];
    for (int t = 0; t < NSAMP; ++t) { bd[t] = 3.0e38f; bi[t] = 0; }
    for (int i = 0; i < NN; ++i) {
        float dx = X[i*3+0]-qx, dy = X[i*3+1]-qy, dz = X[i*3+2]-qz;
        float d  = dx*dx + dy*dy + dz*dz;
        if (d < bd[NSAMP-1]) {
            int p = NSAMP-1;
            while (p > 0 && bd[p-1] > d) { bd[p] = bd[p-1]; bi[p] = bi[p-1]; --p; }
            bd[p] = d; bi[p] = i;
        }
    }
    int fb = bi[0];
    int* g = gidx + (size_t)s * NSAMP;
    for (int t = 0; t < NSAMP; ++t) g[t] = (bd[t] > RAD2) ? fb : bi[t];
}

// ---------------------------------------------------------------------------
// 3) Weight conversion to f16 (K0 padded 67->96) + zero the BN stat buffers.
// ---------------------------------------------------------------------------
__global__ __launch_bounds__(256) void prep_kernel(const float* __restrict__ w0,
                                                   const float* __restrict__ w1,
                                                   const float* __restrict__ w2,
                                                   _Float16* __restrict__ W0h,
                                                   _Float16* __restrict__ W1h,
                                                   _Float16* __restrict__ W2h,
                                                   float* __restrict__ stats /*6*128*/) {
    int tid = threadIdx.x;
    for (int i = tid; i < 64*K0PAD; i += 256) {
        int o = i / K0PAD, c = i % K0PAD;
        W0h[i] = (_Float16)((c < 67) ? w0[o*67 + c] : 0.0f);
    }
    for (int i = tid; i < 64*64;  i += 256) W1h[i] = (_Float16)w1[i];
    for (int i = tid; i < 128*64; i += 256) W2h[i] = (_Float16)w2[i];
    for (int i = tid; i < 6*128;  i += 256) stats[i] = 0.0f;
}

// ---------------------------------------------------------------------------
// 4) Gather + normalize + concat -> x0 f16 [PP x 96]
// ---------------------------------------------------------------------------
__global__ __launch_bounds__(256) void build_x0(const float* __restrict__ xyz,
                                                const float* __restrict__ pts,
                                                const float* __restrict__ newxyz,
                                                const int* __restrict__ gidx,
                                                _Float16* __restrict__ x0) {
    int row = blockIdx.x * 256 + threadIdx.x;   // < PP
    int s   = row / NSAMP;                      // b*SS + s
    int b   = s / SS;
    int pid = gidx[row];
    const float* q  = newxyz + (size_t)s * 3;
    const float* p3 = xyz + ((size_t)b*NN + pid) * 3;
    const float* pf = pts + ((size_t)b*NN + pid) * DD;
    _Float16* dst = x0 + (size_t)row * K0PAD;
    dst[0] = (_Float16)(p3[0]-q[0]);
    dst[1] = (_Float16)(p3[1]-q[1]);
    dst[2] = (_Float16)(p3[2]-q[2]);
    #pragma unroll
    for (int c = 0; c < DD; ++c) dst[3+c] = (_Float16)pf[c];
    #pragma unroll
    for (int c = 3+DD; c < K0PAD; ++c) dst[c] = (_Float16)0.0f;
}

// ---------------------------------------------------------------------------
// 5) WMMA GEMM: Y[PP x O] = X[PP x Kpad] * W^T + bias   (f16 in, f32 acc)
//    Fused per-channel sum / sum-of-squares for batchnorm (LDS ds_add_f32,
//    then one global atomic per channel per block).
//    8 waves/block; wave (wr, wc) owns a 16x16 tile; uniform flow => EXEC=~0.
// ---------------------------------------------------------------------------
__global__ __launch_bounds__(256) void gemm_kernel(const _Float16* __restrict__ X,
                                                   const _Float16* __restrict__ Wh,
                                                   const float* __restrict__ bias,
                                                   _Float16* __restrict__ Y,
                                                   float* __restrict__ gsum,
                                                   float* __restrict__ gsq,
                                                   int Kpad, int O,
                                                   int nColTiles, int rowWaves) {
    __shared__ float ssum[128];
    __shared__ float ssq[128];
    const int tid = threadIdx.x;
    if (tid < 128) { ssum[tid] = 0.0f; ssq[tid] = 0.0f; }
    __syncthreads();

    const int w    = tid >> 5;
    const int lane = tid & 31;
    const int h    = lane >> 4;       // half-wave select
    const int l    = lane & 15;
    const int wc   = w % nColTiles;
    const int wr   = w / nColTiles;
    const int rowBase = blockIdx.x * (16 * rowWaves) + wr * 16;
    const int colBase = wc * 16;

    const _Float16* Xp = X  + (size_t)(rowBase + l) * Kpad;
    const _Float16* Wp = Wh + (size_t)(colBase + l) * Kpad;

    v8f c = {};
    const int nk = Kpad >> 5;
    for (int kt = 0; kt < nk; ++kt) {
        // A fragment: row = rowBase+l, K halves {8h..8h+7} and {16+8h..23+8h}
        const int ka = kt*32 + 8*h;
        v8h a0 = *(const v8h*)(Xp + ka);
        v8h a1 = *(const v8h*)(Xp + ka + 16);
        // B fragment: col = colBase+l, K halves {8h..8h+7} and {16+8h..23+8h}
        const int kb = kt*32 + 8*h;
        v8h b0 = *(const v8h*)(Wp + kb);
        v8h b1 = *(const v8h*)(Wp + kb + 16);
        v16h av, bv;
        #pragma unroll
        for (int i = 0; i < 8; ++i) {
            av[i] = a0[i]; av[i+8] = a1[i];
            bv[i] = b0[i]; bv[i+8] = b1[i];
        }
        c = __builtin_amdgcn_wmma_f32_16x16x32_f16(false, av, false, bv,
                                                   (short)0, c, false, false);
    }

    const int col = colBase + l;
    const float bs = bias[col];
    float ls = 0.0f, lq = 0.0f;
    #pragma unroll
    for (int j = 0; j < 8; ++j) {
        float v = c[j] + bs;                      // row = rowBase + 8h + j
        ls += v; lq += v*v;
        Y[(size_t)(rowBase + 8*h + j) * O + col] = (_Float16)v;
    }
    atomicAdd(&ssum[col], ls);                    // ds_add_f32
    atomicAdd(&ssq[col],  lq);
    __syncthreads();
    if (tid < O) {
        unsafeAtomicAdd(&gsum[tid], ssum[tid]);   // global_atomic_add_f32
        unsafeAtomicAdd(&gsq[tid],  ssq[tid]);
    }
}

// ---------------------------------------------------------------------------
// 6) Fold BN stats into per-channel scale/shift: a = g/sqrt(var+eps),
//    b = beta - a*mean
// ---------------------------------------------------------------------------
__global__ __launch_bounds__(128) void bn_finalize(const float* __restrict__ gsum,
                                                   const float* __restrict__ gsq,
                                                   const float* __restrict__ g,
                                                   const float* __restrict__ beta,
                                                   float* __restrict__ ab, int O) {
    int c = threadIdx.x;
    if (c < O) {
        const float invN = 1.0f / (float)PP;
        float mean = gsum[c] * invN;
        float var  = gsq[c] * invN - mean*mean;
        float inv  = rsqrtf(var + 1.0e-5f);
        float a    = g[c] * inv;
        ab[c]       = a;
        ab[128 + c] = beta[c] - a * mean;
    }
}

// ---------------------------------------------------------------------------
// 7) Normalize + ReLU -> next-layer f16 activations
// ---------------------------------------------------------------------------
__global__ __launch_bounds__(256) void bn_relu(const _Float16* __restrict__ Y,
                                               const float* __restrict__ ab,
                                               _Float16* __restrict__ Xn,
                                               int Omask, int total) {
    int i = blockIdx.x * 256 + threadIdx.x;
    if (i < total) {
        int c = i & Omask;
        float v = ab[c] * (float)Y[i] + ab[128 + c];
        Xn[i] = (_Float16)fmaxf(v, 0.0f);
    }
}

// ---------------------------------------------------------------------------
// 8) Final layer: normalize + ReLU + max over the 32 samples -> output f32
// ---------------------------------------------------------------------------
__global__ __launch_bounds__(256) void maxpool_kernel(const _Float16* __restrict__ Y,
                                                      const float* __restrict__ ab,
                                                      float* __restrict__ outp) {
    int i = blockIdx.x * 256 + threadIdx.x;      // < BB*SS*128
    int c = i & 127;
    int s = i >> 7;
    const _Float16* base = Y + (size_t)s * NSAMP * 128 + c;
    float a = ab[c], b2 = ab[128 + c];
    float m = -3.0e38f;
    #pragma unroll
    for (int j = 0; j < NSAMP; ++j) {
        float v = a * (float)base[j*128] + b2;
        m = fmaxf(m, v);
    }
    outp[i] = fmaxf(m, 0.0f);   // relu before max == relu after max
}

// ---------------------------------------------------------------------------
extern "C" void kernel_launch(void* const* d_in, const int* in_sizes, int n_in,
                              void* d_out, int out_size, void* d_ws, size_t ws_size,
                              hipStream_t stream) {
    const float* xyz  = (const float*)d_in[0];
    const float* pts  = (const float*)d_in[1];
    const float* w0   = (const float*)d_in[2];
    const float* b0   = (const float*)d_in[3];
    const float* g0   = (const float*)d_in[4];
    const float* bb0  = (const float*)d_in[5];
    const float* w1   = (const float*)d_in[6];
    const float* b1   = (const float*)d_in[7];
    const float* g1   = (const float*)d_in[8];
    const float* bb1  = (const float*)d_in[9];
    const float* w2   = (const float*)d_in[10];
    const float* b2   = (const float*)d_in[11];
    const float* g2   = (const float*)d_in[12];
    const float* bb2  = (const float*)d_in[13];

    float* out      = (float*)d_out;
    float* newxyz   = out;                 // first BB*SS*3 floats of the output
    float* newpts   = out + BB*SS*3;

    // workspace carve-up (256-B aligned)
    char* base = (char*)d_ws;
    size_t off = 0;
    auto carve = [&](size_t bytes) -> char* {
        char* p = base + off;
        off = (off + bytes + 255) & ~(size_t)255;
        return p;
    };
    int*      gidx  = (int*)     carve((size_t)PP * sizeof(int));
    _Float16* x0    = (_Float16*)carve((size_t)PP * K0PAD * sizeof(_Float16));
    _Float16* x1    = (_Float16*)carve((size_t)PP * 64    * sizeof(_Float16));
    _Float16* ybuf  = (_Float16*)carve((size_t)PP * 128   * sizeof(_Float16));
    _Float16* W0h   = (_Float16*)carve((size_t)64  * K0PAD * sizeof(_Float16));
    _Float16* W1h   = (_Float16*)carve((size_t)64  * 64    * sizeof(_Float16));
    _Float16* W2h   = (_Float16*)carve((size_t)128 * 64    * sizeof(_Float16));
    float*    stats = (float*)   carve((size_t)6 * 128 * sizeof(float));
    float*    ab    = (float*)   carve((size_t)2 * 128 * sizeof(float));
    (void)ws_size; (void)n_in; (void)in_sizes; (void)out_size;

    float* sum0 = stats;        float* sq0 = stats + 128;
    float* sum1 = stats + 256;  float* sq1 = stats + 384;
    float* sum2 = stats + 512;  float* sq2 = stats + 640;

    // 1) FPS -> new_xyz (directly into d_out)
    fps_kernel<<<BB, 256, 0, stream>>>(xyz, newxyz);
    // 2) ball query -> group indices
    ball_kernel<<<(BB*SS + 127)/128, 128, 0, stream>>>(xyz, newxyz, gidx);
    // 3) weights -> f16, zero stats
    prep_kernel<<<1, 256, 0, stream>>>(w0, w1, w2, W0h, W1h, W2h, stats);
    // 4) gather/concat -> x0
    build_x0<<<PP/256, 256, 0, stream>>>(xyz, pts, newxyz, gidx, x0);

    // layer 0: [PP x 96] x [96 -> 64]
    gemm_kernel<<<PP/32, 256, 0, stream>>>(x0, W0h, b0, ybuf, sum0, sq0, K0PAD, 64, 4, 2);
    bn_finalize<<<1, 128, 0, stream>>>(sum0, sq0, g0, bb0, ab, 64);
    bn_relu<<<(PP*64)/256, 256, 0, stream>>>(ybuf, ab, x1, 63, PP*64);

    // layer 1: [PP x 64] x [64 -> 64]
    gemm_kernel<<<PP/32, 256, 0, stream>>>(x1, W1h, b1, ybuf, sum1, sq1, 64, 64, 4, 2);
    bn_finalize<<<1, 128, 0, stream>>>(sum1, sq1, g1, bb1, ab, 64);
    bn_relu<<<(PP*64)/256, 256, 0, stream>>>(ybuf, ab, x1, 63, PP*64);

    // layer 2: [PP x 64] x [64 -> 128]
    gemm_kernel<<<PP/16, 256, 0, stream>>>(x1, W2h, b2, ybuf, sum2, sq2, 64, 128, 8, 1);
    bn_finalize<<<1, 128, 0, stream>>>(sum2, sq2, g2, bb2, ab, 128);
    // 8) BN + ReLU + max over samples -> new_points
    maxpool_kernel<<<(BB*SS*128)/256, 256, 0, stream>>>(ybuf, ab, newpts);
}
